// GCNClassifier_70188355551726
// MI455X (gfx1250) — matrix operand order repaired and checked by reading, not compiled, not run on hardware
//
#include <hip/hip_runtime.h>
#include <hip/hip_bf16.h>

// ---------------------------------------------------------------------------
// GCN on MI455X (gfx1250): stream fp32 adjacency from HBM, convert to bf16 in
// registers, multiply with v_wmma_f32_16x16x32_bf16 (f32 accumulation).
// Memory-bound: ~1 GB adj traffic -> ~43us floor @ 23.3 TB/s.
// ---------------------------------------------------------------------------

#define N_NODES 8192
#define IN_DIM  32
#define HID_DIM 64
#define OUT_DIM 16

typedef __attribute__((ext_vector_type(16))) __bf16 v16bf;
typedef __attribute__((ext_vector_type(8)))  __bf16 v8bf;
typedef __attribute__((ext_vector_type(8)))  float  v8f;

// ---- WMMA wrapper ----------------------------------------------------------
__device__ __forceinline__ v8f wmma_bf16(v16bf a, v16bf b, v8f c) {
    // 8 args: (neg_a, A, neg_b, B, c_mod, C, reuse_a, reuse_b)
    return __builtin_amdgcn_wmma_f32_16x16x32_bf16(
        false, a, false, b, (short)0, c, false, false);
}

// ---- A-tile: 16x32 fp32 (row-major, ld = N_NODES) -> bf16 WMMA A layout ----
// A as loaded here (contiguous k, same as B): lanes 0-15 hold rows M=0..15; lane-half selects K groups:
//   v0..v7 : K = (lane>>4)*16 + 0..15
__device__ __forceinline__ v16bf load_a_tile(const float* __restrict__ adj,
                                             int row, int k0, int lane) {
    const int khalf = (lane >> 4) * 16;
    const float* p0 = adj + (size_t)row * N_NODES + k0 + khalf;
    const float* p1 = p0 + 8;
    float4 f0 = *(const float4*)(p0);
    float4 f1 = *(const float4*)(p0 + 4);
    float4 f2 = *(const float4*)(p1);
    float4 f3 = *(const float4*)(p1 + 4);
    v16bf a;
    a[0]=(__bf16)f0.x;  a[1]=(__bf16)f0.y;  a[2]=(__bf16)f0.z;  a[3]=(__bf16)f0.w;
    a[4]=(__bf16)f1.x;  a[5]=(__bf16)f1.y;  a[6]=(__bf16)f1.z;  a[7]=(__bf16)f1.w;
    a[8]=(__bf16)f2.x;  a[9]=(__bf16)f2.y;  a[10]=(__bf16)f2.z; a[11]=(__bf16)f2.w;
    a[12]=(__bf16)f3.x; a[13]=(__bf16)f3.y; a[14]=(__bf16)f3.z; a[15]=(__bf16)f3.w;
    return a;
}

// ---- B-tile: 32x16 bf16 from K-major (transposed) matrix T[n][k], ld=8192 --
// B as loaded here (contiguous k, same as A): lane holds column n = lane&15; lanes 0-15: K=k0..k0+15,
// lanes 16-31: K=k0+16..k0+31 (two contiguous 16B loads per lane).
__device__ __forceinline__ v16bf load_b_tile(const __bf16* __restrict__ T,
                                             int n0, int k0, int lane) {
    const int n  = n0 + (lane & 15);
    const int kq = k0 + (lane >> 4) * 16;
    const __bf16* p = T + (size_t)n * N_NODES + kq;
    v8bf lo = *(const v8bf*)(p);
    v8bf hi = *(const v8bf*)(p + 8);
    v16bf b;
#pragma unroll
    for (int i = 0; i < 8; ++i) { b[i] = lo[i]; b[8 + i] = hi[i]; }
    return b;
}

// ---- pack features [8192,32] fp32 -> featT [32][8192] bf16 -----------------
__global__ __launch_bounds__(256)
void pack_features_T(const float* __restrict__ f, __bf16* __restrict__ featT) {
    int idx = blockIdx.x * 256 + threadIdx.x;       // 32*8192 elements
    int n = idx >> 13;                              // /8192
    int k = idx & (N_NODES - 1);
    featT[idx] = (__bf16)f[(size_t)k * IN_DIM + n];
}

// ---------------------------------------------------------------------------
// Layer 1: acc{1,2} = adj{1,2} @ x   (K=8192, N=32), then
//          h1 = relu(acc1@W1+b1) + relu(acc2@W1+b1), written as bf16 h1T[64][8192]
// Block: 256 thr = 8 waves = 2 row-tiles(16 rows) x 4 K-slices. 256 blocks.
// ---------------------------------------------------------------------------
__global__ __launch_bounds__(256)
void gcn_layer1(const float* __restrict__ adj1, const float* __restrict__ adj2,
                const __bf16* __restrict__ featT,
                const float* __restrict__ W1, const float* __restrict__ b1,
                __bf16* __restrict__ h1T) {
    __shared__ float accs[2][32][33];       // [branch][row][col(+pad)]
    __shared__ float w1s[IN_DIM][HID_DIM];
    __shared__ float b1s[HID_DIM];

    const int tid  = threadIdx.x;
    const int lane = tid & 31;
    const int wave = tid >> 5;
    const int rt   = wave & 1;              // row tile in block
    const int ks   = wave >> 1;             // K-slice 0..3
    const int rowBase = blockIdx.x * 32;

    // stage W1/b1, zero accumulation buffer
    for (int i = tid; i < IN_DIM * HID_DIM; i += 256) w1s[i >> 6][i & 63] = W1[i];
    if (tid < HID_DIM) b1s[tid] = b1[tid];
    for (int i = tid; i < 2 * 32 * 33; i += 256) (&accs[0][0][0])[i] = 0.0f;
    __syncthreads();

    v8f acc[2][2] = {};                     // [branch][ntile]
    const int aRow   = rowBase + rt * 16 + (lane & 15);
    const int kBegin = ks * (N_NODES / 4);
    const int kEnd   = kBegin + (N_NODES / 4);

    for (int k0 = kBegin; k0 < kEnd; k0 += 32) {
        // prefetch the stream ~1KB ahead (global_prefetch_b8)
        __builtin_prefetch(adj1 + (size_t)aRow * N_NODES + k0 + 256, 0, 0);
        __builtin_prefetch(adj2 + (size_t)aRow * N_NODES + k0 + 256, 0, 0);

        v16bf a1 = load_a_tile(adj1, aRow, k0, lane);
        v16bf a2 = load_a_tile(adj2, aRow, k0, lane);
        v16bf bt0 = load_b_tile(featT, 0,  k0, lane);
        v16bf bt1 = load_b_tile(featT, 16, k0, lane);
        acc[0][0] = wmma_bf16(a1, bt0, acc[0][0]);
        acc[0][1] = wmma_bf16(a1, bt1, acc[0][1]);
        acc[1][0] = wmma_bf16(a2, bt0, acc[1][0]);
        acc[1][1] = wmma_bf16(a2, bt1, acc[1][1]);
    }

    // C layout: elem j -> row j + (lane>>4)*8, col lane&15 (+16*ntile).
    // K-slices combine via LDS float atomics (ds_add_f32).
    const int mOff = rt * 16 + (lane >> 4) * 8;
    const int cc   = lane & 15;
#pragma unroll
    for (int br = 0; br < 2; ++br)
#pragma unroll
        for (int nt = 0; nt < 2; ++nt)
#pragma unroll
            for (int j = 0; j < 8; ++j)
                atomicAdd(&accs[br][mOff + j][nt * 16 + cc], acc[br][nt][j]);
    __syncthreads();

    // epilogue: 32 rows x 64 cols, shared linear + relu + branch sum -> h1T bf16
    for (int o = tid; o < 32 * HID_DIM; o += 256) {
        const int r = o & 31;               // contiguous rows per column run
        const int c = o >> 5;
        float s1 = b1s[c], s2 = b1s[c];
#pragma unroll
        for (int k = 0; k < IN_DIM; ++k) {
            const float w = w1s[k][c];
            s1 = fmaf(accs[0][r][k], w, s1);
            s2 = fmaf(accs[1][r][k], w, s2);
        }
        const float h = fmaxf(s1, 0.0f) + fmaxf(s2, 0.0f);
        h1T[(size_t)c * N_NODES + rowBase + r] = (__bf16)h;
    }
}

// ---------------------------------------------------------------------------
// Layer 2: acc{1,2} = adj{1,2} @ h1  (K=8192, N=64), then
//          out = relu(acc1@W2+b2) + relu(acc2@W2+b2)   [8192,16] fp32
// ---------------------------------------------------------------------------
__global__ __launch_bounds__(256)
void gcn_layer2(const float* __restrict__ adj1, const float* __restrict__ adj2,
                const __bf16* __restrict__ h1T,
                const float* __restrict__ W2, const float* __restrict__ b2,
                float* __restrict__ out) {
    __shared__ float accs[2][32][65];       // [branch][row][col(+pad)]
    __shared__ float w2s[HID_DIM][OUT_DIM];
    __shared__ float b2s[OUT_DIM];

    const int tid  = threadIdx.x;
    const int lane = tid & 31;
    const int wave = tid >> 5;
    const int rt   = wave & 1;
    const int ks   = wave >> 1;
    const int rowBase = blockIdx.x * 32;

    for (int i = tid; i < HID_DIM * OUT_DIM; i += 256) w2s[i >> 4][i & 15] = W2[i];
    if (tid < OUT_DIM) b2s[tid] = b2[tid];
    for (int i = tid; i < 2 * 32 * 65; i += 256) (&accs[0][0][0])[i] = 0.0f;
    __syncthreads();

    v8f acc[2][4] = {};                     // [branch][ntile], N=64 -> 4 tiles
    const int aRow   = rowBase + rt * 16 + (lane & 15);
    const int kBegin = ks * (N_NODES / 4);
    const int kEnd   = kBegin + (N_NODES / 4);

    for (int k0 = kBegin; k0 < kEnd; k0 += 32) {
        __builtin_prefetch(adj1 + (size_t)aRow * N_NODES + k0 + 256, 0, 0);
        __builtin_prefetch(adj2 + (size_t)aRow * N_NODES + k0 + 256, 0, 0);

        v16bf a1 = load_a_tile(adj1, aRow, k0, lane);
        v16bf a2 = load_a_tile(adj2, aRow, k0, lane);
#pragma unroll
        for (int nt = 0; nt < 4; ++nt) {
            v16bf bt = load_b_tile(h1T, nt * 16, k0, lane);
            acc[0][nt] = wmma_bf16(a1, bt, acc[0][nt]);
            acc[1][nt] = wmma_bf16(a2, bt, acc[1][nt]);
        }
    }

    const int mOff = rt * 16 + (lane >> 4) * 8;
    const int cc   = lane & 15;
#pragma unroll
    for (int br = 0; br < 2; ++br)
#pragma unroll
        for (int nt = 0; nt < 4; ++nt)
#pragma unroll
            for (int j = 0; j < 8; ++j)
                atomicAdd(&accs[br][mOff + j][nt * 16 + cc], acc[br][nt][j]);
    __syncthreads();

    // epilogue: 32 rows x 16 cols -> coalesced fp32 stores
    for (int o = tid; o < 32 * OUT_DIM; o += 256) {
        const int r = o >> 4;
        const int c = o & 15;
        float s1 = b2s[c], s2 = b2s[c];
#pragma unroll
        for (int k = 0; k < HID_DIM; ++k) {
            const float w = w2s[k][c];
            s1 = fmaf(accs[0][r][k], w, s1);
            s2 = fmaf(accs[1][r][k], w, s2);
        }
        out[(size_t)(rowBase + r) * OUT_DIM + c] = fmaxf(s1, 0.0f) + fmaxf(s2, 0.0f);
    }
}

// ---------------------------------------------------------------------------
extern "C" void kernel_launch(void* const* d_in, const int* in_sizes, int n_in,
                              void* d_out, int out_size, void* d_ws, size_t ws_size,
                              hipStream_t stream) {
    const float* adj1     = (const float*)d_in[0];
    const float* adj2     = (const float*)d_in[1];
    const float* features = (const float*)d_in[2];
    const float* W1       = (const float*)d_in[3];
    const float* b1       = (const float*)d_in[4];
    const float* W2       = (const float*)d_in[5];
    const float* b2       = (const float*)d_in[6];
    float* out = (float*)d_out;

    // workspace: featT (32*8192 bf16 = 512KB) then h1T (64*8192 bf16 = 1MB)
    __bf16* featT = (__bf16*)d_ws;
    __bf16* h1T   = featT + (size_t)IN_DIM * N_NODES;

    pack_features_T<<<(IN_DIM * N_NODES) / 256, 256, 0, stream>>>(features, featT);
    gcn_layer1<<<N_NODES / 32, 256, 0, stream>>>(adj1, adj2, featT, W1, b1, h1T);
    gcn_layer2<<<N_NODES / 32, 256, 0, stream>>>(adj1, adj2, h1T, W2, b2, out);
}
